// durationLMLayer2_79559974191175
// MI455X (gfx1250) — matrix operand order repaired and checked by reading, not compiled, not run on hardware
//
#include <hip/hip_runtime.h>

#define Bb 32
#define Ss 256
#define Vv 10000
#define Ee 512
#define Hh 1024
#define Ll 3
#define Mm (Ss*Bb)      // 8192 rows (m = s*B + b)
#define G4H (4*Hh)      // 4096

typedef __attribute__((ext_vector_type(16))) __bf16       bf16x16;
typedef __attribute__((ext_vector_type(8)))  float        f32x8;
typedef __attribute__((ext_vector_type(8)))  unsigned int u32x8;
typedef __attribute__((ext_vector_type(4)))  unsigned int u32x4;
typedef __attribute__((ext_vector_type(4)))  int          i32x4;

__device__ __forceinline__ unsigned short f32_to_bf16(float f) {
  union { float f; unsigned u; } v; v.f = f;
  unsigned u = v.u;
  unsigned r = u + 0x7FFFu + ((u >> 16) & 1u);
  if ((u & 0x7F800000u) == 0x7F800000u) r = u;   // inf/nan: truncate
  return (unsigned short)(r >> 16);
}
__device__ __forceinline__ float bf16_to_f32(unsigned short h) {
  union { unsigned u; float f; } v; v.u = ((unsigned)h) << 16;
  return v.f;
}
__device__ __forceinline__ float sigm(float x) { return 1.0f / (1.0f + __expf(-x)); }

// ---- async global->LDS copy of 16 bytes (gfx1250 async path w/ fallback) ----
__device__ __forceinline__ void async_copy16(unsigned short* lds, const unsigned short* g) {
#if __has_builtin(__builtin_amdgcn_global_load_async_to_lds_b128)
  __builtin_amdgcn_global_load_async_to_lds_b128(
      (__attribute__((address_space(1))) i32x4*)g,
      (__attribute__((address_space(3))) i32x4*)lds, 0, 0);
#else
  *reinterpret_cast<u32x4*>(lds) = *reinterpret_cast<const u32x4*>(g);
#endif
}
__device__ __forceinline__ void async_join() {
#if __has_builtin(__builtin_amdgcn_global_load_async_to_lds_b128)
#if __has_builtin(__builtin_amdgcn_s_wait_asynccnt)
  __builtin_amdgcn_s_wait_asynccnt(0);
#else
  asm volatile("s_wait_asynccnt 0" ::: "memory");
#endif
#endif
}

// ---- WMMA fragment loads (contiguous k order, same for A and B) ----
// A 16x32 bf16: lane L -> M = L%16; lanes<16 K = 0..15, lanes>=16 K = 16..31
__device__ __forceinline__ bf16x16 load_frag_a(const unsigned short* __restrict__ base,
                                               int ld, int m0, int k0, int lane) {
  const unsigned short* p = base + (size_t)(m0 + (lane & 15)) * ld + k0 + ((lane & 16) ? 16 : 0);
  u32x4 lo = *reinterpret_cast<const u32x4*>(p);
  u32x4 hi = *reinterpret_cast<const u32x4*>(p + 8);
  u32x8 v;
  v[0]=lo[0]; v[1]=lo[1]; v[2]=lo[2]; v[3]=lo[3];
  v[4]=hi[0]; v[5]=hi[1]; v[6]=hi[2]; v[7]=hi[3];
  return __builtin_bit_cast(bf16x16, v);
}
// B 32x16 bf16 where B[k][n] = W[n][k], W row-major [N,K]:
// lane L -> N = n0 + L%16 ; K = k0 + (L<16 ? 0..15 : 16..31)  => 32 contiguous bytes
__device__ __forceinline__ bf16x16 load_frag_b(const unsigned short* __restrict__ base,
                                               int ld, int n0, int k0, int lane) {
  const unsigned short* p = base + (size_t)(n0 + (lane & 15)) * ld + k0 + ((lane & 16) ? 16 : 0);
  u32x8 v = *reinterpret_cast<const u32x8*>(p);
  return __builtin_bit_cast(bf16x16, v);
}
__device__ __forceinline__ f32x8 wmma_bf16(bf16x16 a, bf16x16 b, f32x8 c) {
  return __builtin_amdgcn_wmma_f32_16x16x32_bf16(false, a, false, b, (short)0, c, false, false);
}

// ---------------- small elementwise kernels ----------------
__global__ void cvt_bf16_kernel(const float* __restrict__ in,
                                unsigned short* __restrict__ out, size_t n) {
  size_t i = (size_t)blockIdx.x * blockDim.x + threadIdx.x;
  if (i < n) out[i] = f32_to_bf16(in[i]);
}

// layer-norm over seq dim per batch row + mask/gate for each layer
__global__ __launch_bounds__(256) void prep_kernel(const float* __restrict__ fix_pred,
                                                   float* __restrict__ maskW,
                                                   float* __restrict__ gateW) {
  int b = blockIdx.x;             // 0..31
  int s = threadIdx.x;            // 0..255
  float x = fix_pred[b * Ss + s];
  float sum = x, sq = x * x;
  for (int o = 16; o > 0; o >>= 1) {
    sum += __shfl_down(sum, o);
    sq  += __shfl_down(sq, o);
  }
  __shared__ float red[20];
  int lane = threadIdx.x & 31, wv = threadIdx.x >> 5;
  if (lane == 0) { red[wv] = sum; red[wv + 8] = sq; }
  __syncthreads();
  if (threadIdx.x == 0) {
    float ts = 0.f, tq = 0.f;
    for (int i = 0; i < 8; i++) { ts += red[i]; tq += red[i + 8]; }
    float mu = ts / 256.0f;
    float var = tq / 256.0f - mu * mu;
    red[16] = mu; red[17] = rsqrtf(var + 1e-5f);
  }
  __syncthreads();
  float fp = (x - red[16]) * red[17];
  fp = (fp + 1.96f) / 3.92f * 12.0f;
  float alpha0 = (11.0f - fp) * 0.25f;     // RESCALE = 12//3 = 4
  for (int l = 0; l < Ll; l++) {
    float a = alpha0 - (float)l;
    float m = sigm(a);
    float g = sigm(a + 1.0f) - m;
    maskW[l * Mm + s * Bb + b] = m;
    gateW[l * Mm + s * Bb + b] = g;
  }
}

// gather embedding rows (as bf16) in m = s*B + b order
__global__ __launch_bounds__(128) void embed_kernel(const int* __restrict__ src,
                                                    const float* __restrict__ emb,
                                                    unsigned short* __restrict__ Xemb) {
  int m = blockIdx.x;                 // 0..8191
  int s = m >> 5, b = m & 31;
  int tok = src[b * Ss + s];
  const float* e = emb + (size_t)tok * Ee;
  unsigned short* o = Xemb + (size_t)m * Ee;
  for (int j = threadIdx.x; j < Ee; j += 128) o[j] = f32_to_bf16(e[j]);
}

// inp_seq = x_prev*(1-gate) + input_x*gate (gate scalar per row m)
__global__ __launch_bounds__(256) void blend_kernel(const unsigned short* __restrict__ xprev,
                                                    const unsigned short* __restrict__ xin,
                                                    const float* __restrict__ gateL,
                                                    unsigned short* __restrict__ out) {
  int m = blockIdx.x;
  float g = gateL[m];
  const unsigned short* ir = xin + (size_t)m * Hh;
  const unsigned short* xr = xprev ? xprev + (size_t)m * Hh : nullptr;
  unsigned short* o = out + (size_t)m * Hh;
  for (int j = threadIdx.x; j < Hh; j += 256) {
    float xv = xr ? bf16_to_f32(xr[j]) : 0.0f;
    float iv = bf16_to_f32(ir[j]);
    o[j] = f32_to_bf16(xv * (1.0f - g) + iv * g);
  }
}

__global__ void init_state_kernel(const float* __restrict__ h0l, const float* __restrict__ c0l,
                                  unsigned short* __restrict__ hbf0, float* __restrict__ hf,
                                  float* __restrict__ cf, unsigned int* __restrict__ sync) {
  int i = blockIdx.x * blockDim.x + threadIdx.x;
  if (i == 0) *sync = 0u;
  if (i < Bb * Hh) {
    float h = h0l[i], c = c0l[i];
    hf[i] = h; cf[i] = c;
    hbf0[i] = f32_to_bf16(h);
  }
}

// ---------------- generic bf16 WMMA GEMM: C = A[M,K] @ W[N,K]^T (+bias) ----------------
// block tile 128(M) x 64(N), 8 waves (4x2), wave tile 32x32 = 4 WMMA accumulators.
// A/B k-slices staged in LDS via async global->LDS copies.
__global__ __launch_bounds__(256) void gemm_bf16_kernel(
    const unsigned short* __restrict__ A, int ldA,
    const unsigned short* __restrict__ W, int ldW,
    const float* __restrict__ bias0, const float* __restrict__ bias1,
    void* __restrict__ Cout, int N, int K,
    int outMode /*0:f32 rowmajor, 1:bf16 rowmajor, 2:f32 [b,s] permute*/,
    int act /*0:none, 1:tanh*/) {
  __shared__ unsigned short shA[128 * 32];   // 8 KB
  __shared__ unsigned short shB[64 * 32];    // 4 KB
  const int tid  = threadIdx.x;
  const int lane = tid & 31;
  const int wv   = __builtin_amdgcn_readfirstlane(tid) >> 5;  // scalar wave id 0..7
  const int wm = wv & 3, wn = wv >> 2;
  const int m_blk = blockIdx.y * 128;
  const int n_blk = blockIdx.x * 64;
  const int n_base = n_blk + wn * 32;
  const bool nv0 = (n_base < N);            // scalar (wave-uniform, SGPR) predicates
  const bool nv1 = (n_base + 16 < N);

  const int arow = tid >> 2, acc_ = (tid & 3) * 8;  // staging coords
  const int brow = tid >> 2, bcc  = (tid & 3) * 8;
  const bool bvalid = (n_blk + brow < N);

  f32x8 acc[2][2] = {};
  for (int k = 0; k < K; k += 32) {
    __syncthreads();                       // previous iter's LDS reads done
    // stage A 128x32 (two 16B chunks per thread) and B 64x32 (one chunk)
    async_copy16(&shA[arow * 32 + acc_],
                 A + (size_t)(m_blk + arow) * ldA + k + acc_);
    async_copy16(&shA[(arow + 64) * 32 + acc_],
                 A + (size_t)(m_blk + arow + 64) * ldA + k + acc_);
    if (bvalid)
      async_copy16(&shB[brow * 32 + bcc],
                   W + (size_t)(n_blk + brow) * ldW + k + bcc);
    async_join();
    __syncthreads();

    bf16x16 a0 = load_frag_a(shA, 32, wm * 32, 0, lane);
    bf16x16 a1 = load_frag_a(shA, 32, wm * 32 + 16, 0, lane);
    if (nv0) {
      bf16x16 b0 = load_frag_b(shB, 32, wn * 32, 0, lane);
      acc[0][0] = wmma_bf16(a0, b0, acc[0][0]);
      acc[1][0] = wmma_bf16(a1, b0, acc[1][0]);
    }
    if (nv1) {
      bf16x16 b1 = load_frag_b(shB, 32, wn * 32 + 16, 0, lane);
      acc[0][1] = wmma_bf16(a0, b1, acc[0][1]);
      acc[1][1] = wmma_bf16(a1, b1, acc[1][1]);
    }
  }

  for (int nt = 0; nt < 2; nt++) {
    int n0 = n_base + nt * 16;
    if (n0 >= N) continue;
    int ncol = n0 + (lane & 15);
    float bb = (bias0 ? bias0[ncol] : 0.0f) + (bias1 ? bias1[ncol] : 0.0f);
    for (int mt = 0; mt < 2; mt++) {
      int m0 = m_blk + wm * 32 + mt * 16;
      f32x8 c = acc[mt][nt];
#pragma unroll
      for (int r = 0; r < 8; r++) {
        int row = m0 + ((lane & 16) ? 8 + r : r);
        float v = c[r] + bb;
        if (act == 1) v = tanhf(v);
        if (outMode == 0) {
          ((float*)Cout)[(size_t)row * N + ncol] = v;
        } else if (outMode == 1) {
          ((unsigned short*)Cout)[(size_t)row * N + ncol] = f32_to_bf16(v);
        } else {
          int b = row & 31, s = row >> 5;                 // row = s*B + b
          ((float*)Cout)[((size_t)b * Ss + s) * N + ncol] = v;
        }
      }
    }
  }
}

// ---------------- persistent weight-stationary LSTM scan (one layer) ----------------
// 64 blocks x 4 waves. Wave = (column tile jt = blockIdx.x, gate = wave id).
// Each wave keeps its 16x1024 W_hh slice resident in VGPRs as 32 pre-formatted
// B fragments (256 VGPRs) -> zero weight traffic inside the time loop.
__global__ __launch_bounds__(128, 1) void lstm_scan_kernel(
    const float* __restrict__ G,            // [S*B, 4H] input-side pre-activations (+biases)
    const unsigned short* __restrict__ Whh, // [4H, H] bf16
    const float* __restrict__ maskL,        // [S*B]
    unsigned short* __restrict__ hbuf,      // ping-pong: 2 x [B,H] bf16
    float* __restrict__ hf,                 // [B,H] f32
    float* __restrict__ cf,                 // [B,H] f32
    unsigned short* __restrict__ xout,      // [S*B, H] bf16 (layer output)
    unsigned int* __restrict__ sync,
    int nblocks) {
  const int lane = threadIdx.x & 31;
  const int gate = __builtin_amdgcn_readfirstlane(threadIdx.x) >> 5;  // 0..3 scalar
  const int jt   = blockIdx.x;              // 0..63
  const int n0   = jt * 16;
  const int ncol = n0 + (lane & 15);
  const int rb   = (lane & 16) ? 8 : 0;

  // resident weight fragments for this (gate, column-tile): W rows gate*H+n0..+15
  bf16x16 wfrag[32];
#pragma unroll
  for (int kk = 0; kk < 32; kk++)
    wfrag[kk] = load_frag_b(Whh, Hh, gate * Hh + n0, kk * 32, lane);

  __shared__ float gsh[4][Bb][16];          // gate grids meet here
  __shared__ float msh[Bb];

  for (int t = 0; t < Ss; ++t) {
    if (threadIdx.x < Bb) msh[threadIdx.x] = maskL[t * Bb + threadIdx.x];

    const unsigned short* hrd = hbuf + (size_t)(t & 1) * Bb * Hh;
    unsigned short*       hwr = hbuf + (size_t)((t + 1) & 1) * Bb * Hh;

    // init accumulators from input-side pre-activations
    f32x8 acc[2];
    const float* Gt = G + (size_t)(t * Bb) * G4H + gate * Hh + ncol;
#pragma unroll
    for (int mt = 0; mt < 2; mt++)
#pragma unroll
      for (int r = 0; r < 8; r++)
        acc[mt][r] = Gt[(size_t)(mt * 16 + rb + r) * G4H];

    // recurrent h @ Whh^T for this gate/tile (weights already in VGPRs)
#pragma unroll
    for (int kk = 0; kk < 32; kk++) {
      bf16x16 a0 = load_frag_a(hrd, Hh, 0, kk * 32, lane);
      bf16x16 a1 = load_frag_a(hrd, Hh, 16, kk * 32, lane);
      acc[0] = wmma_bf16(a0, wfrag[kk], acc[0]);
      acc[1] = wmma_bf16(a1, wfrag[kk], acc[1]);
    }

    // publish this wave's gate grid [32 x 16] to LDS
#pragma unroll
    for (int mt = 0; mt < 2; mt++)
#pragma unroll
      for (int r = 0; r < 8; r++)
        gsh[gate][mt * 16 + rb + r][lane & 15] = acc[mt][r];
    __syncthreads();

    // pointwise LSTM cell + adaptive skip blend: 512 cells, 4 per thread
#pragma unroll
    for (int c = 0; c < 4; c++) {
      int cell = c * 128 + threadIdx.x;     // coalesced
      int br   = cell >> 4;                 // 0..31
      int col  = cell & 15;
      int j    = n0 + col;
      size_t idx = (size_t)br * Hh + j;
      float gi = gsh[0][br][col], gfv = gsh[1][br][col];
      float gg = gsh[2][br][col], go  = gsh[3][br][col];
      float c_old = cf[idx];
      float c_new = sigm(gfv) * c_old + sigm(gi) * tanhf(gg);
      float h_new = sigm(go) * tanhf(c_new);
      float m = msh[br];
      c_new = c_new * (1.0f - m) + c_old * m;
      float h_old = hf[idx];
      h_new = h_new * (1.0f - m) + h_old * m;
      cf[idx] = c_new;
      hf[idx] = h_new;
      unsigned short hb = f32_to_bf16(h_new);
      hwr[idx] = hb;
      xout[(size_t)(t * Bb + br) * Hh + j] = hb;
    }

    // grid-wide sync: release writes, arrive, spin-acquire
    __threadfence();
    __syncthreads();
    if (threadIdx.x == 0) {
      __hip_atomic_fetch_add(sync, 1u, __ATOMIC_ACQ_REL, __HIP_MEMORY_SCOPE_AGENT);
      unsigned target = (unsigned)nblocks * (unsigned)(t + 1);
      while (__hip_atomic_load(sync, __ATOMIC_ACQUIRE, __HIP_MEMORY_SCOPE_AGENT) < target)
        __builtin_amdgcn_s_sleep(2);
    }
    __syncthreads();
  }
}

// ---------------- host orchestration ----------------
extern "C" void kernel_launch(void* const* d_in, const int* in_sizes, int n_in,
                              void* d_out, int out_size, void* d_ws, size_t ws_size,
                              hipStream_t stream) {
  const int* src        = (const int*)d_in[0];
  const float* fix_pred = (const float*)d_in[1];
  const float* emb      = (const float*)d_in[2];
  const float* W_lin    = (const float*)d_in[3];
  const float* b_lin    = (const float*)d_in[4];
  const float* W_ih     = (const float*)d_in[5];
  const float* W_hh     = (const float*)d_in[6];
  const float* b_ih     = (const float*)d_in[7];
  const float* b_hh     = (const float*)d_in[8];
  const float* fc_W1    = (const float*)d_in[9];
  const float* fc_b1    = (const float*)d_in[10];
  const float* fc_W2    = (const float*)d_in[11];
  const float* fc_b2    = (const float*)d_in[12];
  const float* h0       = (const float*)d_in[13];
  const float* c0       = (const float*)d_in[14];

  char* ws = (char*)d_ws;
  size_t off = 0;
  auto alloc = [&](size_t bytes) -> char* {
    char* p = ws + off;
    off = (off + bytes + 255) & ~(size_t)255;
    return p;
  };
  float*          maskW   = (float*)alloc((size_t)Ll * Mm * 4);
  float*          gateW   = (float*)alloc((size_t)Ll * Mm * 4);
  unsigned short* Wlin_b  = (unsigned short*)alloc((size_t)Hh * Ee * 2);
  unsigned short* Wih_b   = (unsigned short*)alloc((size_t)Ll * G4H * Hh * 2);
  unsigned short* Whh_b   = (unsigned short*)alloc((size_t)Ll * G4H * Hh * 2);
  unsigned short* W1_b    = (unsigned short*)alloc((size_t)Ee * Hh * 2);
  unsigned short* W2_b    = (unsigned short*)alloc((size_t)Vv * Ee * 2);
  unsigned short* Xemb    = (unsigned short*)alloc((size_t)Mm * Ee * 2);
  unsigned short* Xproj   = (unsigned short*)alloc((size_t)Mm * Hh * 2);  // input_x bf16
  unsigned short* Xa      = (unsigned short*)alloc((size_t)Mm * Hh * 2);  // layer output x
  unsigned short* Inp     = (unsigned short*)alloc((size_t)Mm * Hh * 2);  // blended inp_seq
  float*          Gws     = (float*)alloc((size_t)Mm * G4H * 4);          // pre-activations
  unsigned short* Hid     = (unsigned short*)alloc((size_t)Mm * Ee * 2);  // tanh(fc1) bf16
  unsigned short* hbuf    = (unsigned short*)alloc((size_t)2 * Bb * Hh * 2);
  float*          hf      = (float*)alloc((size_t)Bb * Hh * 4);
  float*          cf      = (float*)alloc((size_t)Bb * Hh * 4);
  unsigned int*   sync    = (unsigned int*)alloc(256);

  auto cvt = [&](const float* in, unsigned short* out, size_t n) {
    cvt_bf16_kernel<<<(unsigned)((n + 255) / 256), 256, 0, stream>>>(in, out, n);
  };
  cvt(W_lin, Wlin_b, (size_t)Hh * Ee);
  cvt(W_ih,  Wih_b,  (size_t)Ll * G4H * Hh);
  cvt(W_hh,  Whh_b,  (size_t)Ll * G4H * Hh);
  cvt(fc_W1, W1_b,   (size_t)Ee * Hh);
  cvt(fc_W2, W2_b,   (size_t)Vv * Ee);

  prep_kernel<<<Bb, Ss, 0, stream>>>(fix_pred, maskW, gateW);
  embed_kernel<<<Mm, 128, 0, stream>>>(src, emb, Xemb);

  // input projection: [8192,512] @ [1024,512]^T + b_lin -> bf16 [8192,1024]
  gemm_bf16_kernel<<<dim3(Hh / 64, Mm / 128), 256, 0, stream>>>(
      Xemb, Ee, Wlin_b, Ee, b_lin, nullptr, Xproj, Hh, Ee, 1, 0);

  for (int l = 0; l < Ll; l++) {
    const unsigned short* xprev = (l == 0) ? nullptr : Xa;
    blend_kernel<<<Mm, 256, 0, stream>>>(xprev, Xproj, gateW + (size_t)l * Mm, Inp);
    // input-side pre-activations: [8192,1024] @ [4096,1024]^T + (b_ih+b_hh) -> f32
    gemm_bf16_kernel<<<dim3(G4H / 64, Mm / 128), 256, 0, stream>>>(
        Inp, Hh, Wih_b + (size_t)l * G4H * Hh, Hh,
        b_ih + (size_t)l * G4H, b_hh + (size_t)l * G4H, Gws, G4H, Hh, 0, 0);
    init_state_kernel<<<(Bb * Hh + 255) / 256, 256, 0, stream>>>(
        h0 + (size_t)l * Bb * Hh, c0 + (size_t)l * Bb * Hh, hbuf, hf, cf, sync);
    lstm_scan_kernel<<<64, 128, 0, stream>>>(
        Gws, Whh_b + (size_t)l * G4H * Hh, maskW + (size_t)l * Mm,
        hbuf, hf, cf, Xa, sync, 64);
  }

  // FC1: tanh([8192,1024] @ [512,1024]^T + b1) -> bf16
  gemm_bf16_kernel<<<dim3(Ee / 64, Mm / 128), 256, 0, stream>>>(
      Xa, Hh, W1_b, Hh, fc_b1, nullptr, Hid, Ee, Hh, 1, 1);
  // FC2: [8192,512] @ [10000,512]^T + b2 -> f32, row-major [S*B,V]
  gemm_bf16_kernel<<<dim3((Vv + 63) / 64, Mm / 128), 256, 0, stream>>>(
      Hid, Ee, W2_b, Ee, fc_b2, nullptr, d_out, Vv, Ee, 0, 2);
}
